// STAGIN_36009005809711
// MI455X (gfx1250) — compile-verified, matrix operand order checked by reading, not run on hardware
//
#include <hip/hip_runtime.h>
#include <math.h>

#define B_ 4
#define W_ 32
#define N_ 400
#define NP_ 416             // N padded to multiple of 32 (K padding for WMMA)
#define D_ 128
#define C_ 2
#define L_ 4
#define T_ 512
#define TW_ 16              // T / W
#define BW_ (B_*W_)         // 128
#define BWN_ (BW_*N_)       // 51200
#define HTOT_ ((long long)BWN_*D_)   // 6,553,600

typedef __bf16 v16bf __attribute__((ext_vector_type(16)));
typedef float  v8f   __attribute__((ext_vector_type(8)));
typedef unsigned short ushort_t;

union FragU { v16bf v; unsigned u[8]; };

__device__ __forceinline__ ushort_t f2bf(float f) {
  unsigned u = __builtin_bit_cast(unsigned, f);
  u = (u + 0x7FFFu + ((u >> 16) & 1u)) >> 16;   // round-to-nearest-even
  return (ushort_t)u;
}

// ---------------------------------------------------------------------------
// Batched GEMM, C = A(MxK) * Bt(NcxK)^T (+bias).  A,Bt pre-converted bf16 with
// K a multiple of 32 (zero padded).  Branchless inner loop: per lane the
// fragment is two contiguous 16-byte runs -> 4 x global_load_b128 + 1 WMMA.
// Block = 128 threads (4 waves), each wave owns one 16x16 C tile.
// ---------------------------------------------------------------------------
__global__ void gemm_bf16_nt(const ushort_t* __restrict__ A, const ushort_t* __restrict__ Bt,
                             const float* __restrict__ bias, float* __restrict__ C,
                             int M, int Nc, int K, int lda, int ldb, int ldc,
                             long long sA, long long sB, long long sC) {
  const int lane = threadIdx.x & 31;
  const int wave = threadIdx.x >> 5;
  const int m0 = blockIdx.x * 64 + wave * 16;
  const int n0 = blockIdx.y * 16;
  const int khi = (lane >> 4) * 8;          // hi half-wave covers K+8 (ISA layout)
  const int arow = min(m0 + (lane & 15), M - 1);   // clamp: branchless OOB handling
  const int brow = min(n0 + (lane & 15), Nc - 1);

  const ushort_t* Ap = A + (long long)blockIdx.z * sA + (long long)arow * lda + khi;
  const ushort_t* Bp = Bt + (long long)blockIdx.z * sB + (long long)brow * ldb + khi;

  v8f acc = {};
  for (int kc = 0; kc < K; kc += 32) {
    FragU fa, fb;
    *(uint4*)&fa.u[0] = *(const uint4*)(Ap + kc);        // k in [khi, khi+8)
    *(uint4*)&fa.u[4] = *(const uint4*)(Ap + kc + 16);   // k in [khi+16, khi+24)
    *(uint4*)&fb.u[0] = *(const uint4*)(Bp + kc);
    *(uint4*)&fb.u[4] = *(const uint4*)(Bp + kc + 16);
    acc = __builtin_amdgcn_wmma_f32_16x16x32_bf16(false, fa.v, false, fb.v,
                                                  (short)0, acc, false, false);
  }

  float* Cb = C + (long long)blockIdx.z * sC;
  const int col = n0 + (lane & 15);
  const int rb  = m0 + ((lane >> 4) ? 8 : 0);
  float bv = 0.f;
  if (bias && col < Nc) bv = bias[col];
#pragma unroll
  for (int t = 0; t < 8; ++t) {
    int row = rb + t;
    if (row < M && col < Nc) Cb[(long long)row * ldc + col] = acc[t] + bv;
  }
}

// ---------------------------------------------------------------------------
// f32 -> bf16 conversion with K zero-padding:  dst[r,0:Kp] = bf16(src[r,0:Kv])
// ---------------------------------------------------------------------------
__global__ void conv_pad_bf16(const float* __restrict__ src, int srcld,
                              ushort_t* __restrict__ dst, int Kv, int Kp,
                              long long rows) {
  long long i = (long long)blockIdx.x * blockDim.x + threadIdx.x;
  if (i >= rows * Kp) return;
  long long r = i / Kp; int k = (int)(i - r * Kp);
  float v = (k < Kv) ? src[r * (long long)srcld + k] : 0.f;
  dst[i] = f2bf(v);
}

// h f32 [BW,N,D] -> hT bf16 [BW,D,NP] (zero padded in n), LDS-tiled transpose
__global__ void transpose_bf16(const float* __restrict__ h, ushort_t* __restrict__ hT) {
  __shared__ float tile[32][33];
  const int bw = blockIdx.z;
  const int n0 = blockIdx.x * 32;       // 13 blocks -> 416
  const int d0 = blockIdx.y * 32;       // 4 blocks -> 128
  const int tx = threadIdx.x, ty = threadIdx.y;   // (32, 8)
  const float* hb = h + (long long)bw * N_ * D_;
  for (int yy = ty; yy < 32; yy += 8) {
    int n = n0 + yy;
    tile[yy][tx] = (n < N_) ? hb[(long long)n * D_ + d0 + tx] : 0.f;
  }
  __syncthreads();
  ushort_t* ht = hT + (long long)bw * D_ * NP_;
  for (int yy = ty; yy < 32; yy += 8) {
    ht[(long long)(d0 + yy) * NP_ + n0 + tx] = f2bf(tile[tx][yy]);
  }
}

// ---------------------------------------------------------------------------
// GRU recurrence (input projection precomputed by WMMA GEMM).
// ---------------------------------------------------------------------------
__global__ void gru_scan(const float* __restrict__ gi_all,   // [T,B,3D]
                         const float* __restrict__ whh,      // [3D,D]
                         const float* __restrict__ bhh,      // [3D]
                         float* __restrict__ enc) {          // [W,B,D]
  __shared__ float hs[B_ * D_];
  __shared__ float ghs[B_ * 3 * D_];
  const int tid = threadIdx.x;
  hs[tid] = 0.f;
  __syncthreads();
  for (int t = 0; t < T_; ++t) {
    const float* gi = gi_all + (long long)t * (B_ * 3 * D_);
    for (int u = tid; u < B_ * 3 * D_; u += 512) {
      int b = u / (3 * D_), j = u % (3 * D_);
      const float* hrow = hs + b * D_;
      const float* wrow = whh + (long long)j * D_;
      float s = bhh[j];
      for (int d2 = 0; d2 < D_; ++d2) s += hrow[d2] * wrow[d2];
      ghs[u] = s;
    }
    __syncthreads();
    {
      const int b = tid / D_, d = tid % D_;
      const float* gib = gi + b * 3 * D_;
      const float* ghb = ghs + b * 3 * D_;
      float r = 1.f / (1.f + expf(-(gib[d]          + ghb[d])));
      float z = 1.f / (1.f + expf(-(gib[D_ + d]     + ghb[D_ + d])));
      float n = tanhf(gib[2 * D_ + d] + r * ghb[2 * D_ + d]);
      float hn = (1.f - z) * n + z * hs[tid];
      hs[tid] = hn;                       // own element only
      if (((t + 1) & (TW_ - 1)) == 0)
        enc[((long long)(t / TW_) * B_ + b) * D_ + d] = hn;
    }
    __syncthreads();
  }
}

// encproj[b,w,d] = init_b[d] + sum_j enc[w,b,j] * init_w[d, 400+j]
__global__ void encproj_kernel(const float* __restrict__ enc, const float* __restrict__ init_w,
                               const float* __restrict__ init_b, float* __restrict__ ep) {
  int bw = blockIdx.x, b = bw / W_, w = bw % W_, d = threadIdx.x;
  const float* e  = enc + ((long long)w * B_ + b) * D_;
  const float* wr = init_w + (long long)d * (N_ + D_) + N_;
  float s = init_b[d];
  for (int j = 0; j < D_; ++j) s += e[j] * wr[j];
  ep[(long long)bw * D_ + d] = s;
}

// h += broadcast(ep); also emit bf16 copy of h
__global__ void add_encproj_bf(float* __restrict__ h, const float* __restrict__ ep,
                               ushort_t* __restrict__ hbf, long long total) {
  long long i = (long long)blockIdx.x * blockDim.x + threadIdx.x;
  if (i >= total) return;
  int d = (int)(i & (D_ - 1));
  long long bw = i / ((long long)N_ * D_);
  float v = h[i] + ep[bw * D_ + d];
  h[i] = v;
  hbf[i] = f2bf(v);
}

// zin_bf16 = bf16(agg + eps[l] * h)
__global__ void eps_add_bf(const float* __restrict__ agg, const float* __restrict__ h,
                           const float* __restrict__ eps, int l,
                           ushort_t* __restrict__ zin, long long total) {
  long long i = (long long)blockIdx.x * blockDim.x + threadIdx.x;
  if (i < total) zin[i] = f2bf(agg[i] + eps[l] * h[i]);
}

// ---------------------------------------------------------------------------
// BatchNorm over rows (train mode, biased variance)
// ---------------------------------------------------------------------------
__global__ void bn_stats(const float* __restrict__ y, float* __restrict__ acc, int R) {
  int j = threadIdx.x;                          // 128 columns
  long long r0 = (long long)blockIdx.x * 256;
  float s = 0.f, q = 0.f;
  for (int i = 0; i < 256; ++i) {
    long long r = r0 + i;
    if (r < R) { float v = y[r * D_ + j]; s += v; q += v * v; }
  }
  atomicAdd(&acc[j], s);
  atomicAdd(&acc[D_ + j], q);
}

// x = relu(BN(yin)); optionally write f32 (yout) and/or bf16 (obf)
__global__ void bn_apply_relu(const float* __restrict__ yin, const float* __restrict__ acc,
                              const float* __restrict__ g, const float* __restrict__ be,
                              float* __restrict__ yout, ushort_t* __restrict__ obf,
                              long long total, float invR) {
  long long i = (long long)blockIdx.x * blockDim.x + threadIdx.x;
  if (i >= total) return;
  int j = (int)(i & (D_ - 1));
  float m = acc[j] * invR;
  float v = acc[D_ + j] * invR - m * m;
  float x = (yin[i] - m) * rsqrtf(v + 1e-5f) * g[j] + be[j];
  x = x > 0.f ? x : 0.f;
  if (yout) yout[i] = x;
  if (obf)  obf[i] = f2bf(x);
}

// ---------------------------------------------------------------------------
// SERO readout
// ---------------------------------------------------------------------------
__global__ void mean_nodes(const float* __restrict__ h, float* __restrict__ m) {
  int bw = blockIdx.x, d = threadIdx.x;
  const float* p = h + (long long)bw * N_ * D_ + d;
  float s = 0.f;
  for (int n = 0; n < N_; ++n) s += p[(long long)n * D_];
  m[(long long)bw * D_ + d] = s * (1.f / N_);
}

__global__ void lin_act(const float* __restrict__ X, const float* __restrict__ Wt,
                        const float* __restrict__ bias, float* __restrict__ Y,
                        int K, int Nout, int act) {
  int r = blockIdx.x, j = threadIdx.x;
  if (j >= Nout) return;
  const float* x = X + (long long)r * K;
  const float* w = Wt + (long long)j * K;
  float s = bias ? bias[j] : 0.f;
  for (int k = 0; k < K; ++k) s += x[k] * w[k];
  if (act == 1)      s = fmaxf(s, 0.f);
  else if (act == 2) s = 1.f / (1.f + expf(-s));
  Y[(long long)r * Nout + j] = s;
}

__global__ void bn_gelu_128(float* __restrict__ y, const float* __restrict__ g,
                            const float* __restrict__ be) {
  int j = threadIdx.x;
  float s = 0.f, q = 0.f;
  for (int r = 0; r < BW_; ++r) { float v = y[r * D_ + j]; s += v; q += v * v; }
  float m = s * (1.f / BW_);
  float inv = rsqrtf(q * (1.f / BW_) - m * m + 1e-5f);
  for (int r = 0; r < BW_; ++r) {
    float x = (y[r * D_ + j] - m) * inv * g[j] + be[j];
    y[r * D_ + j] = 0.5f * x * (1.f + erff(x * 0.70710678118654752f));
  }
}

__global__ void sero_graph(const float* __restrict__ h, const float* __restrict__ nattn,
                           float* __restrict__ graph) {
  int bw = blockIdx.x, d = threadIdx.x;
  const float* p = h + (long long)bw * N_ * D_;
  const float* a = nattn + (long long)bw * N_;
  float s = 0.f;
  for (int n = 0; n < N_; ++n) s += p[(long long)n * D_ + d] * a[n];
  graph[(long long)bw * D_ + d] = s * (1.f / N_);
}

// ---------------------------------------------------------------------------
// Temporal transformer (tiny: W=32, B=4)
// ---------------------------------------------------------------------------
__global__ void attn_scores(const float* __restrict__ qkv, float* __restrict__ aw) {
  __shared__ float sc[W_][W_ + 1];
  int b = blockIdx.x;
  int i = threadIdx.x >> 5, j = threadIdx.x & 31;
  const float* q = qkv + ((long long)(b * W_ + i)) * 384;
  const float* k = qkv + ((long long)(b * W_ + j)) * 384 + 128;
  float s = 0.f;
  for (int d = 0; d < D_; ++d) s += q[d] * k[d];
  s *= 0.08838834764831845f;   // 1/sqrt(128)
  sc[i][j] = s;
  __syncthreads();
  float mx = -1e30f;
  for (int t = 0; t < W_; ++t) mx = fmaxf(mx, sc[i][t]);
  float e = expf(s - mx);
  __syncthreads();
  sc[i][j] = e;
  __syncthreads();
  float sum = 0.f;
  for (int t = 0; t < W_; ++t) sum += sc[i][t];
  aw[((long long)(b * W_ + i)) * W_ + j] = e / sum;
}

__global__ void attn_ctx(const float* __restrict__ aw, const float* __restrict__ qkv,
                         float* __restrict__ ctx) {
  int r = blockIdx.x;            // b*W + i
  int b = r / W_;
  int d = threadIdx.x;
  const float* a = aw + (long long)r * W_;
  float s = 0.f;
  for (int j = 0; j < W_; ++j)
    s += a[j] * qkv[((long long)(b * W_ + j)) * 384 + 256 + d];
  ctx[(long long)r * D_ + d] = s;
}

__global__ void lin_ln(const float* __restrict__ X, const float* __restrict__ Wt,
                       const float* __restrict__ bias, const float* __restrict__ resid,
                       const float* __restrict__ g, const float* __restrict__ be,
                       float* __restrict__ Y, int K) {
  __shared__ float red[D_];
  int r = blockIdx.x, j = threadIdx.x;
  const float* x = X + (long long)r * K;
  const float* w = Wt + (long long)j * K;
  float s = bias[j];
  for (int k = 0; k < K; ++k) s += x[k] * w[k];
  if (resid) s += resid[(long long)r * D_ + j];
  red[j] = s; __syncthreads();
  for (int st = 64; st > 0; st >>= 1) { if (j < st) red[j] += red[j + st]; __syncthreads(); }
  float mean = red[0] * (1.f / D_);
  __syncthreads();
  float d = s - mean;
  red[j] = d * d; __syncthreads();
  for (int st = 64; st > 0; st >>= 1) { if (j < st) red[j] += red[j + st]; __syncthreads(); }
  float var = red[0] * (1.f / D_);
  Y[(long long)r * D_ + j] = d * rsqrtf(var + 1e-5f) * g[j] + be[j];
}

__global__ void pool_cls(const float* __restrict__ att2, const float* __restrict__ clsw,
                         const float* __restrict__ clsb, float* __restrict__ lat,
                         float* __restrict__ logits) {
  __shared__ float p[D_];
  int b = blockIdx.x, d = threadIdx.x;
  float s = 0.f;
  for (int i = 0; i < W_; ++i) s += att2[((long long)(b * W_ + i)) * D_ + d];
  lat[b * D_ + d] = s;
  p[d] = s; __syncthreads();
  if (d < C_) {
    float t = clsb[d];
    for (int k = 0; k < D_; ++k) t += p[k] * clsw[d * D_ + k];
    atomicAdd(&logits[b * C_ + d], t);
  }
}

// ---------------------------------------------------------------------------
// Orthogonality penalty
// ---------------------------------------------------------------------------
__global__ void row_max(const float* __restrict__ Mb, float* __restrict__ rmax) {
  __shared__ float red[128];
  long long row = blockIdx.x;
  const float* p = Mb + row * N_;
  int j = threadIdx.x;
  float mx = -1e30f;
  for (int t = j; t < N_; t += 128) mx = fmaxf(mx, p[t]);
  red[j] = mx; __syncthreads();
  for (int st = 64; st > 0; st >>= 1) { if (j < st) red[j] = fmaxf(red[j], red[j + st]); __syncthreads(); }
  if (j == 0) rmax[row] = red[0];
}

__global__ void ortho_sumsq(const float* __restrict__ Mb, const float* __restrict__ rmax,
                            float* __restrict__ persq) {
  __shared__ float red[256];
  int bw = blockIdx.x, tid = threadIdx.x;
  const float* Mp = Mb   + (long long)bw * N_ * N_;
  const float* rm = rmax + (long long)bw * N_;
  float s = 0.f;
  for (int idx = tid; idx < N_ * N_; idx += 256) {
    int n = idx / N_, m2 = idx % N_;
    if (m2 >= n) {
      float v = Mp[idx] / rm[n] - (m2 == n ? 1.f : 0.f);
      s += v * v;
    }
  }
  red[tid] = s; __syncthreads();
  for (int st = 128; st > 0; st >>= 1) { if (tid < st) red[tid] += red[tid + st]; __syncthreads(); }
  if (tid == 0) persq[bw] = red[0];
}

__global__ void ortho_final(const float* __restrict__ persq, float* __restrict__ ortho) {
  __shared__ float red[128];
  int j = threadIdx.x;
  red[j] = sqrtf(persq[j]);
  __syncthreads();
  for (int st = 64; st > 0; st >>= 1) { if (j < st) red[j] += red[j + st]; __syncthreads(); }
  if (j == 0) ortho[0] += red[0] * (1.f / BW_);
}

// ---------------------------------------------------------------------------
extern "C" void kernel_launch(void* const* d_in, const int* in_sizes, int n_in,
                              void* d_out, int out_size, void* d_ws, size_t ws_size,
                              hipStream_t stream) {
  (void)in_sizes; (void)n_in; (void)ws_size;
  const float* node_identity = (const float*)d_in[0];
  const float* adjacency     = (const float*)d_in[1];
  const float* t_seq         = (const float*)d_in[2];
  // d_in[3] endpoints: deterministic (i+1)*16, folded into gru_scan
  const float* gru_wih = (const float*)d_in[4];
  const float* gru_whh = (const float*)d_in[5];
  const float* gru_bih = (const float*)d_in[6];
  const float* gru_bhh = (const float*)d_in[7];
  const float* init_w  = (const float*)d_in[8];
  const float* init_b  = (const float*)d_in[9];
  const float* gin_eps = (const float*)d_in[10];
  const float* gin_w1  = (const float*)d_in[11];
  const float* gin_b1  = (const float*)d_in[12];
  const float* gin_g1  = (const float*)d_in[13];
  const float* gin_be1 = (const float*)d_in[14];
  const float* gin_w2  = (const float*)d_in[15];
  const float* gin_b2  = (const float*)d_in[16];
  const float* gin_g2  = (const float*)d_in[17];
  const float* gin_be2 = (const float*)d_in[18];
  const float* sero_we = (const float*)d_in[19];
  const float* sero_eb = (const float*)d_in[20];
  const float* sero_g  = (const float*)d_in[21];
  const float* sero_bt = (const float*)d_in[22];
  const float* sero_wa = (const float*)d_in[23];
  const float* sero_ba = (const float*)d_in[24];
  const float* attn_wqkv = (const float*)d_in[25];
  const float* attn_bqkv = (const float*)d_in[26];
  const float* attn_wo   = (const float*)d_in[27];
  const float* attn_bo   = (const float*)d_in[28];
  const float* ln1_g = (const float*)d_in[29];
  const float* ln1_b = (const float*)d_in[30];
  const float* ln2_g = (const float*)d_in[31];
  const float* ln2_b = (const float*)d_in[32];
  const float* mlp_w1 = (const float*)d_in[33];
  const float* mlp_b1 = (const float*)d_in[34];
  const float* mlp_w2 = (const float*)d_in[35];
  const float* mlp_b2 = (const float*)d_in[36];
  const float* cls_w  = (const float*)d_in[37];
  const float* cls_b  = (const float*)d_in[38];

  // ----- workspace layout -----
  float* ws = (float*)d_ws;
  long long o = 0;
  float* bufh   = ws + o;  o += HTOT_;
  float* bufa   = ws + o;  o += HTOT_;
  float* bufb   = ws + o;  o += HTOT_;
  float* gi_all = ws + o;  o += (long long)T_ * B_ * 3 * D_;
  float* enc    = ws + o;  o += (long long)W_ * B_ * D_;
  float* ep     = ws + o;  o += (long long)BW_ * D_;
  float* mvec   = ws + o;  o += (long long)BW_ * D_;
  float* emb    = ws + o;  o += (long long)BW_ * D_;
  float* graph  = ws + o;  o += (long long)BW_ * D_;
  float* qkv    = ws + o;  o += (long long)BW_ * 384;
  float* ctx    = ws + o;  o += (long long)BW_ * D_;
  float* att    = ws + o;  o += (long long)BW_ * D_;
  float* mh     = ws + o;  o += (long long)BW_ * 256;
  float* att2   = ws + o;  o += (long long)BW_ * D_;
  float* bnacc  = ws + o;  o += 256;
  float* rmax   = ws + o;  o += (long long)BWN_;
  float* persq  = ws + o;  o += BW_;
  float* Mbuf   = ws + o;  o += (long long)BW_ * N_ * N_;
  // bf16 region (carved as float-pairs, all 16B aligned)
  ushort_t* bigAbf  = (ushort_t*)(ws + o);  o += ((long long)BWN_ * NP_ + 1) / 2;   // node_identity, then adjacency
  ushort_t* hTbf    = (ushort_t*)(ws + o);  o += ((long long)BW_ * D_ * NP_ + 1) / 2;
  ushort_t* hbf     = (ushort_t*)(ws + o);  o += (HTOT_ + 1) / 2;
  ushort_t* zybf    = (ushort_t*)(ws + o);  o += (HTOT_ + 1) / 2;                   // zin, then activated y1
  ushort_t* tseqbf  = (ushort_t*)(ws + o);  o += ((long long)T_ * B_ * NP_ + 1) / 2;
  ushort_t* wihbf   = (ushort_t*)(ws + o);  o += ((long long)3 * D_ * NP_ + 1) / 2;
  ushort_t* initwbf = (ushort_t*)(ws + o);  o += ((long long)D_ * NP_ + 1) / 2;
  ushort_t* w1bf    = (ushort_t*)(ws + o);  o += ((long long)L_ * D_ * D_ + 1) / 2;
  ushort_t* w2bf    = (ushort_t*)(ws + o);  o += ((long long)L_ * D_ * D_ + 1) / 2;

  // ----- output layout -----
  float* out        = (float*)d_out;
  float* out_logits = out;
  float* out_nattn  = out + 8;
  float* out_taw    = out_nattn + (long long)L_ * BWN_;
  float* out_lat    = out_taw   + (long long)L_ * B_ * W_ * W_;
  float* out_ortho  = out_lat   + (long long)L_ * B_ * D_;
  hipMemsetAsync(d_out, 0, (size_t)out_size * sizeof(float), stream);

  const int eltBlocks = (int)((HTOT_ + 255) / 256);
  const int padBlocks = (int)(((long long)BWN_ * NP_ + 255) / 256);

  // ===== one-time operand conversions (f32 -> zero-padded bf16) =====
  conv_pad_bf16<<<(T_*B_*NP_+255)/256, 256, 0, stream>>>(t_seq, N_, tseqbf, N_, NP_, T_*B_);
  conv_pad_bf16<<<(3*D_*NP_+255)/256, 256, 0, stream>>>(gru_wih, N_, wihbf, N_, NP_, 3*D_);
  conv_pad_bf16<<<(D_*NP_+255)/256, 256, 0, stream>>>(init_w, N_+D_, initwbf, N_, NP_, D_);
  conv_pad_bf16<<<(L_*D_*D_+255)/256, 256, 0, stream>>>(gin_w1, D_, w1bf, D_, D_, L_*D_);
  conv_pad_bf16<<<(L_*D_*D_+255)/256, 256, 0, stream>>>(gin_w2, D_, w2bf, D_, D_, L_*D_);
  conv_pad_bf16<<<padBlocks, 256, 0, stream>>>(node_identity, N_, bigAbf, N_, NP_, BWN_);

  // ===== GRU encoder =====
  gemm_bf16_nt<<<dim3(32, 24, 1), 128, 0, stream>>>(
      tseqbf, wihbf, gru_bih, gi_all, T_*B_, 3*D_, NP_, NP_, NP_, 3*D_, 0, 0, 0);
  gru_scan<<<1, 512, 0, stream>>>(gi_all, gru_whh, gru_bhh, enc);
  encproj_kernel<<<BW_, D_, 0, stream>>>(enc, init_w, init_b, ep);

  // ===== init projection: h0 = node_identity @ init_w[:, :400]^T + ep =====
  gemm_bf16_nt<<<dim3(BWN_/64, D_/16, 1), 128, 0, stream>>>(
      bigAbf, initwbf, nullptr, bufh, BWN_, D_, NP_, NP_, NP_, D_, 0, 0, 0);
  add_encproj_bf<<<eltBlocks, 256, 0, stream>>>(bufh, ep, hbf, HTOT_);
  transpose_bf16<<<dim3(NP_/32, D_/32, BW_), dim3(32, 8), 0, stream>>>(bufh, hTbf);

  // now reuse bigAbf for adjacency (init GEMM already consumed node_identity)
  conv_pad_bf16<<<padBlocks, 256, 0, stream>>>(adjacency, N_, bigAbf, N_, NP_, BWN_);

  float* hcur = bufh;
  float* tmpa = bufa;

  for (int l = 0; l < L_; ++l) {
    // ----- GIN: agg = adjacency @ h  == adj_bf16 (NT) hT_bf16, batched -----
    gemm_bf16_nt<<<dim3(7, D_/16, BW_), 128, 0, stream>>>(
        bigAbf, hTbf, nullptr, tmpa, N_, D_, NP_, NP_, NP_, D_,
        (long long)N_ * NP_, (long long)D_ * NP_, (long long)N_ * D_);
    eps_add_bf<<<eltBlocks, 256, 0, stream>>>(tmpa, hcur, gin_eps, l, zybf, HTOT_);

    // MLP1 + BN + ReLU (bf16 activation out only)
    gemm_bf16_nt<<<dim3(BWN_/64, D_/16, 1), 128, 0, stream>>>(
        zybf, w1bf + (long long)l * D_ * D_, gin_b1 + l * D_, bufb,
        BWN_, D_, D_, D_, D_, D_, 0, 0, 0);
    hipMemsetAsync(bnacc, 0, 256 * sizeof(float), stream);
    bn_stats<<<BWN_/256, D_, 0, stream>>>(bufb, bnacc, BWN_);
    bn_apply_relu<<<eltBlocks, 256, 0, stream>>>(
        bufb, bnacc, gin_g1 + l * D_, gin_be1 + l * D_, nullptr, zybf, HTOT_, 1.f/BWN_);

    // MLP2 + BN + ReLU -> new h (f32 into tmpa, bf16 into hbf)
    gemm_bf16_nt<<<dim3(BWN_/64, D_/16, 1), 128, 0, stream>>>(
        zybf, w2bf + (long long)l * D_ * D_, gin_b2 + l * D_, bufb,
        BWN_, D_, D_, D_, D_, D_, 0, 0, 0);
    hipMemsetAsync(bnacc, 0, 256 * sizeof(float), stream);
    bn_stats<<<BWN_/256, D_, 0, stream>>>(bufb, bnacc, BWN_);
    bn_apply_relu<<<eltBlocks, 256, 0, stream>>>(
        bufb, bnacc, gin_g2 + l * D_, gin_be2 + l * D_, tmpa, hbf, HTOT_, 1.f/BWN_);
    { float* t = hcur; hcur = tmpa; tmpa = t; }
    transpose_bf16<<<dim3(NP_/32, D_/32, BW_), dim3(32, 8), 0, stream>>>(hcur, hTbf);

    // ----- SERO readout -----
    float* nattn_l = out_nattn + (long long)l * BWN_;
    mean_nodes<<<BW_, D_, 0, stream>>>(hcur, mvec);
    lin_act<<<BW_, D_, 0, stream>>>(mvec, sero_we + (long long)l * D_ * D_,
                                    sero_eb + l * D_, emb, D_, D_, 0);
    bn_gelu_128<<<1, D_, 0, stream>>>(emb, sero_g + l * D_, sero_bt + l * D_);
    lin_act<<<BW_, 416, 0, stream>>>(emb, sero_wa + (long long)l * N_ * D_,
                                     sero_ba + l * N_, nattn_l, D_, N_, 2);
    sero_graph<<<BW_, D_, 0, stream>>>(hcur, nattn_l, graph);

    // ----- temporal transformer -----
    float* aw_l = out_taw + (long long)l * B_ * W_ * W_;
    lin_act<<<BW_, 384, 0, stream>>>(graph, attn_wqkv + (long long)l * 384 * D_,
                                     attn_bqkv + l * 384, qkv, D_, 384, 0);
    attn_scores<<<B_, 1024, 0, stream>>>(qkv, aw_l);
    attn_ctx<<<BW_, D_, 0, stream>>>(aw_l, qkv, ctx);
    lin_ln<<<BW_, D_, 0, stream>>>(ctx, attn_wo + (long long)l * D_ * D_,
                                   attn_bo + l * D_, nullptr,
                                   ln1_g + l * D_, ln1_b + l * D_, att, D_);
    lin_act<<<BW_, 256, 0, stream>>>(att, mlp_w1 + (long long)l * 256 * D_,
                                     mlp_b1 + l * 256, mh, D_, 256, 1);
    lin_ln<<<BW_, D_, 0, stream>>>(mh, mlp_w2 + (long long)l * D_ * 256,
                                   mlp_b2 + l * D_, att,
                                   ln2_g + l * D_, ln2_b + l * D_, att2, 256);
    pool_cls<<<B_, D_, 0, stream>>>(att2, cls_w + (long long)l * C_ * D_,
                                    cls_b + l * C_, out_lat + (long long)l * B_ * D_,
                                    out_logits);

    // ----- orthogonality penalty: M = h h^T per (b,w), WMMA NT batched -----
    gemm_bf16_nt<<<dim3(7, 25, BW_), 128, 0, stream>>>(
        hbf, hbf, nullptr, Mbuf, N_, N_, D_, D_, D_, N_,
        (long long)N_ * D_, (long long)N_ * D_, (long long)N_ * N_);
    row_max<<<BWN_, 128, 0, stream>>>(Mbuf, rmax);
    ortho_sumsq<<<BW_, 256, 0, stream>>>(Mbuf, rmax, persq);
    ortho_final<<<1, 128, 0, stream>>>(persq, out_ortho);
  }
}
